// MultiHeadAttention_67027259621527
// MI455X (gfx1250) — matrix operand order repaired and checked by reading, not compiled, not run on hardware
//
#include <hip/hip_runtime.h>

// ---------------------------------------------------------------------------
// MI455X (gfx1250) multi-head attention: bf16 WMMA GEMMs (64x64 wave tiles) +
// fused RoPE + flash attention with TDM (tensor_load_to_lds) double-buffered
// K/V staging.  B=4, T=1024, DIM=2048, H=16, HD=128.
// ---------------------------------------------------------------------------

typedef __bf16 v16bf __attribute__((ext_vector_type(16)));
typedef float  v8f   __attribute__((ext_vector_type(8)));
typedef unsigned u32x4 __attribute__((ext_vector_type(4)));
typedef int      i32x8 __attribute__((ext_vector_type(8)));
typedef int      i32x4 __attribute__((ext_vector_type(4)));

#define WMMA_BF16(a, b, c) \
  __builtin_amdgcn_wmma_f32_16x16x32_bf16(false, (a), false, (b), (short)0, (c), false, false)

static __device__ __forceinline__ unsigned short f2bf(float f) {
  unsigned u = __float_as_uint(f);
  u += 0x7FFFu + ((u >> 16) & 1u);
  return (unsigned short)(u >> 16);
}

union FragU {
  v16bf v;
  unsigned u[8];
  unsigned short s[16];
};

// A-operand fragment (16x32 bf16, this file's contiguous k order, not the ISA table order): lane = row (m = lane&15),
// half = lane>>4; VGPR p holds k-pair half*16 + p*2.
static __device__ __forceinline__ v16bf load_frag_a(const unsigned* __restrict__ dw,
                                                    int row0, int k0, int ld_dw) {
  const int lane = threadIdx.x & 31;
  const int m = lane & 15, half = lane >> 4;
  FragU r;
#pragma unroll
  for (int p = 0; p < 8; ++p) {
    int k = half * 16 + p * 2;
    r.u[p] = dw[(size_t)(row0 + m) * ld_dw + ((unsigned)(k0 + k) >> 1)];
  }
  return r.v;
}

// B-operand fragment (32x16 bf16): lane = column n; lanes 0-15 hold K=0..15,
// lanes 16-31 hold K=16..31.  Source row-major with K contiguous
// (weight rows / K rows): row0+n selects the source row.
static __device__ __forceinline__ v16bf load_frag_b(const unsigned* __restrict__ dw,
                                                    int row0, int k0, int ld_dw) {
  const int lane = threadIdx.x & 31;
  const int n = lane & 15, half = lane >> 4;
  FragU r;
#pragma unroll
  for (int p = 0; p < 8; ++p) {
    int k = half * 16 + p * 2;
    r.u[p] = dw[(size_t)(row0 + n) * ld_dw + ((unsigned)(k0 + k) >> 1)];
  }
  return r.v;
}

// B-operand fragment gathered from a row-major [K][N] LDS tile (k strided):
// element e of lane (n, half) is src[(half*16+e)*ld + n0 + n].
static __device__ __forceinline__ v16bf load_frag_b_rowk(const unsigned short* __restrict__ src,
                                                         int n0, int ld) {
  const int lane = threadIdx.x & 31;
  const int n = lane & 15, half = lane >> 4;
  FragU r;
#pragma unroll
  for (int e = 0; e < 16; ++e) {
    int k = half * 16 + e;
    r.s[e] = src[k * ld + n0 + n];
  }
  return r.v;
}

// ---------------------------------------------------------------------------
// TDM: 2D tile load Global -> LDS via tensor descriptor (D#).
// data_size = 2 bytes; tile = rows x rowlen elements; row stride in elements.
// Toolchain uses the 6-arg builtin: (g0, g1, g2, g3, g_extra, cpol).
// ---------------------------------------------------------------------------
static __device__ __forceinline__ void tdm_load_2d(unsigned lds_off, const void* gaddr,
                                                   int rows, int rowlen, int row_stride) {
  unsigned long long ga = (unsigned long long)(uintptr_t)gaddr;
  u32x4 g0;
  g0[0] = 1u;                                                  // count=1 (valid), user mode
  g0[1] = lds_off;                                             // lds_addr (bytes)
  g0[2] = (unsigned)(ga & 0xFFFFFFFFu);                        // global_addr[31:0]
  g0[3] = (unsigned)((ga >> 32) & 0x1FFFFFFu) | (2u << 30);    // global_addr[56:32] | type=2
  i32x8 g1;
  g1[0] = 1 << 16;                                             // data_size=1 (2 bytes)
  g1[1] = (rowlen & 0xFFFF) << 16;                             // tensor_dim0 lo16
  g1[2] = ((unsigned)rowlen >> 16) | ((rows & 0xFFFF) << 16);  // tensor_dim0 hi | tensor_dim1 lo
  g1[3] = ((unsigned)rows >> 16) | ((rowlen & 0xFFFF) << 16);  // tensor_dim1 hi | tile_dim0
  g1[4] = rows & 0xFFFF;                                       // tile_dim1 (tile_dim2=0)
  g1[5] = row_stride;                                          // tensor_dim0_stride lo32
  g1[6] = 0;                                                   // stride hi | dim1_stride lo
  g1[7] = 0;
  i32x4 gz4 = {0, 0, 0, 0};
  i32x8 gz8 = {0, 0, 0, 0, 0, 0, 0, 0};
  __builtin_amdgcn_tensor_load_to_lds(g0, g1, gz4, gz4, gz8, 0);
}

// ---------------------------------------------------------------------------
// fp32 -> bf16 conversion
// ---------------------------------------------------------------------------
__global__ void cvt_f32_to_bf16(const float4* __restrict__ in,
                                ushort4* __restrict__ out, int n4) {
  int i = blockIdx.x * blockDim.x + threadIdx.x;
  if (i < n4) {
    float4 v = in[i];
    ushort4 o;
    o.x = f2bf(v.x); o.y = f2bf(v.y); o.z = f2bf(v.z); o.w = f2bf(v.w);
    out[i] = o;
  }
}

// ---------------------------------------------------------------------------
// C = A(MxK) * W(NxK)^T, bf16 in, f32 accumulate.  Each wave owns a 64x64
// tile (4x4 accumulators, 16 WMMA per k-step, 16 b128 loads per k-step);
// block of 4 waves covers 128x128.
// ---------------------------------------------------------------------------
__global__ __launch_bounds__(128)
void gemm_bf16_wmma(const unsigned short* __restrict__ A,
                    const unsigned short* __restrict__ W,
                    unsigned short* __restrict__ outb,
                    float* __restrict__ outf,
                    const float* __restrict__ fcos,
                    const float* __restrict__ fsin,
                    int K, int N, int rope) {
  const unsigned* Adw = (const unsigned*)A;
  const unsigned* Wdw = (const unsigned*)W;
  const int wave = threadIdx.x >> 5, lane = threadIdx.x & 31;
  const int m0 = blockIdx.y * 128 + (wave >> 1) * 64;
  const int n0 = blockIdx.x * 128 + (wave & 1) * 64;
  const int lddw = K >> 1;

  const v8f z = {0.f, 0.f, 0.f, 0.f, 0.f, 0.f, 0.f, 0.f};
  v8f acc[4][4];
#pragma unroll
  for (int mi = 0; mi < 4; ++mi)
#pragma unroll
    for (int ni = 0; ni < 4; ++ni) acc[mi][ni] = z;

  for (int k0 = 0; k0 < K; k0 += 32) {
    v16bf a[4], b[4];
#pragma unroll
    for (int i = 0; i < 4; ++i) a[i] = load_frag_a(Adw, m0 + i * 16, k0, lddw);
#pragma unroll
    for (int i = 0; i < 4; ++i) b[i] = load_frag_b(Wdw, n0 + i * 16, k0, lddw);
#pragma unroll
    for (int mi = 0; mi < 4; ++mi)
#pragma unroll
      for (int ni = 0; ni < 4; ++ni)
        acc[mi][ni] = WMMA_BF16(a[mi], b[ni], acc[mi][ni]);
  }

  const int half = lane >> 4, col_lo = lane & 15;
#pragma unroll
  for (int mi = 0; mi < 4; ++mi) {
#pragma unroll
    for (int ni = 0; ni < 4; ++ni) {
#pragma unroll
      for (int j = 0; j < 8; ++j) {
        int row = m0 + mi * 16 + j + half * 8;      // global M (= b*T + t)
        int col = n0 + ni * 16 + col_lo;            // global N
        float v = acc[mi][ni][j];
        if (rope) {
          float part = __shfl_xor(v, 1, 32);        // partner column (col^1)
          int t  = row & 1023;
          int fi = (col & 127) >> 1;
          float c = fcos[t * 64 + fi];
          float s = fsin[t * 64 + fi];
          v = (col & 1) ? (part * s + v * c) : (v * c - part * s);
        }
        if (outb) outb[(size_t)row * N + col] = f2bf(v);
        else      outf[(size_t)row * N + col] = v;
      }
    }
  }
}

// ---------------------------------------------------------------------------
// Flash attention.  grid = (T/64, H, B), block = 128 (4 waves, 16 q rows ea).
// K/V chunks (32x128 bf16) staged in LDS by the Tensor Data Mover, double-
// buffered: wave 0 issues chunk c+1's pair, waits TENSORcnt<=2 (chunk c's
// pair complete, c+1's in flight), then the block barrier publishes LDS.
// ---------------------------------------------------------------------------
__global__ __launch_bounds__(128)
void attn_wmma(const unsigned short* __restrict__ qb,
               const unsigned short* __restrict__ kb,
               const unsigned short* __restrict__ vb,
               unsigned short* __restrict__ ob) {
  __shared__ unsigned short Kb[2][32 * 128];   // [buf][kv][hd] row-major
  __shared__ unsigned short Vb[2][32 * 128];   // [buf][kv][hd] row-major
  __shared__ unsigned short Pb[4][16 * 32];    // per-wave P tile [q][kv]

  const int tid = threadIdx.x, wave = tid >> 5, lane = tid & 31;
  const int h = blockIdx.y, b = blockIdx.z;
  const int q0 = blockIdx.x * 64 + wave * 16;
  const int rowbase = b * 1024;
  const int half = lane >> 4, nlo = lane & 15;

  const unsigned ldsK = (unsigned)(uintptr_t)&Kb[0][0];
  const unsigned ldsV = (unsigned)(uintptr_t)&Vb[0][0];

  const unsigned* qdw = (const unsigned*)qb + h * 64;   // head col offset (dw)

  v16bf qf[4];
#pragma unroll
  for (int kk = 0; kk < 4; ++kk)
    qf[kk] = load_frag_a(qdw, rowbase + q0, kk * 32, 1024);

  const v8f z = {0.f, 0.f, 0.f, 0.f, 0.f, 0.f, 0.f, 0.f};
  v8f of[8] = {z, z, z, z, z, z, z, z};
  float m8[8], l8[8];
#pragma unroll
  for (int j = 0; j < 8; ++j) { m8[j] = -1e30f; l8[j] = 0.f; }

  const float scale = 0.08838834764831845f;  // 1/sqrt(128)

  // prologue: DMA chunk 0 into buffer 0
  if (wave == 0) {
    size_t goff = (size_t)rowbase * 2048 + h * 128;
    tdm_load_2d(ldsK, kb + goff, 32, 128, 2048);
    tdm_load_2d(ldsV, vb + goff, 32, 128, 2048);
  }

  for (int c = 0; c < 32; ++c) {
    if (wave == 0) {
      if (c + 1 < 32) {
        size_t goff = ((size_t)(rowbase + (c + 1) * 32)) * 2048 + h * 128;
        unsigned boff = (unsigned)((c + 1) & 1) * (32 * 128 * 2);
        tdm_load_2d(ldsK + boff, kb + goff, 32, 128, 2048);
        tdm_load_2d(ldsV + boff, vb + goff, 32, 128, 2048);
        __builtin_amdgcn_s_wait_tensorcnt(2);   // chunk c's pair done
      } else {
        __builtin_amdgcn_s_wait_tensorcnt(0);
      }
    }
    __syncthreads();

    const unsigned short* Kc = Kb[c & 1];
    const unsigned short* Vc = Vb[c & 1];

    // ---- S = (Q K^T) * scale ----
    v8f s0 = z, s1 = z;
    const unsigned* Ktdw = (const unsigned*)Kc;
#pragma unroll
    for (int kk = 0; kk < 4; ++kk) {
      v16bf bk0 = load_frag_b(Ktdw, 0,  kk * 32, 64);
      v16bf bk1 = load_frag_b(Ktdw, 16, kk * 32, 64);
      s0 = WMMA_BF16(qf[kk], bk0, s0);
      s1 = WMMA_BF16(qf[kk], bk1, s1);
    }

    // ---- online softmax ----
#pragma unroll
    for (int j = 0; j < 8; ++j) {
      float x0 = s0[j] * scale, x1 = s1[j] * scale;
      float cm = fmaxf(x0, x1);
#pragma unroll
      for (int d = 1; d < 16; d <<= 1) cm = fmaxf(cm, __shfl_xor(cm, d, 32));
      float mnew  = fmaxf(m8[j], cm);
      float alpha = __expf(m8[j] - mnew);
      float p0 = __expf(x0 - mnew), p1 = __expf(x1 - mnew);
      float rs = p0 + p1;
#pragma unroll
      for (int d = 1; d < 16; d <<= 1) rs += __shfl_xor(rs, d, 32);
      l8[j] = l8[j] * alpha + rs;
      m8[j] = mnew;
#pragma unroll
      for (int nt = 0; nt < 8; ++nt) of[nt][j] *= alpha;
      int prow = j + half * 8;
      Pb[wave][prow * 32 + nlo]      = f2bf(p0);
      Pb[wave][prow * 32 + 16 + nlo] = f2bf(p1);
    }

    // ---- O += P (16x32) @ V (32x128), V gathered from row-major LDS ----
    v16bf pa = load_frag_a((const unsigned*)Pb[wave], 0, 0, 16);
#pragma unroll
    for (int nt = 0; nt < 8; ++nt) {
      v16bf bv = load_frag_b_rowk(Vc, nt * 16, 128);
      of[nt] = WMMA_BF16(pa, bv, of[nt]);
    }
    __syncthreads();
  }

  // ---- epilogue: O /= l ----
#pragma unroll
  for (int nt = 0; nt < 8; ++nt) {
#pragma unroll
    for (int j = 0; j < 8; ++j) {
      int row = rowbase + q0 + j + half * 8;
      int col = h * 128 + nt * 16 + nlo;
      ob[(size_t)row * 2048 + col] = f2bf(of[nt][j] / l8[j]);
    }
  }
}

// ---------------------------------------------------------------------------
// host launcher
// ---------------------------------------------------------------------------
extern "C" void kernel_launch(void* const* d_in, const int* in_sizes, int n_in,
                              void* d_out, int out_size, void* d_ws, size_t ws_size,
                              hipStream_t stream) {
  const float* x  = (const float*)d_in[0];
  const float* fc = (const float*)d_in[1];
  const float* fs = (const float*)d_in[2];
  const float* wq = (const float*)d_in[3];
  const float* wk = (const float*)d_in[4];
  const float* wv = (const float*)d_in[5];
  const float* wo = (const float*)d_in[6];

  char* ws = (char*)d_ws;
  const size_t MB = 1u << 20;
  unsigned short* xb  = (unsigned short*)(ws + 0);
  unsigned short* wqb = (unsigned short*)(ws + 16 * MB);
  unsigned short* wkb = (unsigned short*)(ws + 24 * MB);
  unsigned short* wvb = (unsigned short*)(ws + 32 * MB);
  unsigned short* wob = (unsigned short*)(ws + 40 * MB);
  unsigned short* qb  = (unsigned short*)(ws + 48 * MB);
  unsigned short* kbf = (unsigned short*)(ws + 64 * MB);
  unsigned short* vbf = (unsigned short*)(ws + 80 * MB);
  unsigned short* obf = (unsigned short*)(ws + 96 * MB);

  cvt_f32_to_bf16<<<8192, 256, 0, stream>>>((const float4*)x,  (ushort4*)xb,  2097152);
  cvt_f32_to_bf16<<<4096, 256, 0, stream>>>((const float4*)wq, (ushort4*)wqb, 1048576);
  cvt_f32_to_bf16<<<4096, 256, 0, stream>>>((const float4*)wk, (ushort4*)wkb, 1048576);
  cvt_f32_to_bf16<<<4096, 256, 0, stream>>>((const float4*)wv, (ushort4*)wvb, 1048576);
  cvt_f32_to_bf16<<<4096, 256, 0, stream>>>((const float4*)wo, (ushort4*)wob, 1048576);

  dim3 gg(16, 32), gb(128);
  gemm_bf16_wmma<<<gg, gb, 0, stream>>>(xb, wqb, qb,  nullptr, fc, fs, 2048, 2048, 1);
  gemm_bf16_wmma<<<gg, gb, 0, stream>>>(xb, wkb, kbf, nullptr, fc, fs, 2048, 2048, 1);
  gemm_bf16_wmma<<<gg, gb, 0, stream>>>(xb, wvb, vbf, nullptr, nullptr, nullptr, 2048, 2048, 0);

  attn_wmma<<<dim3(16, 16, 4), 128, 0, stream>>>(qb, kbf, vbf, obf);

  gemm_bf16_wmma<<<gg, gb, 0, stream>>>(obf, wob, nullptr, (float*)d_out,
                                        nullptr, nullptr, 2048, 2048, 0);
}
